// GPTLanguageModel_43877385896276
// MI455X (gfx1250) — matrix operand order repaired and checked by reading, not compiled, not run on hardware
//
#include <hip/hip_runtime.h>

// ---------------- CDNA5 WMMA types ----------------
typedef __attribute__((ext_vector_type(16))) __bf16 v16bf;
typedef __attribute__((ext_vector_type(8)))  __bf16 v8bf;
typedef __attribute__((ext_vector_type(8)))  float  v8f;

union V16U { v16bf v; v8bf h[2]; };

// ---------------- model constants ----------------
constexpr int Lc  = 4;
constexpr int Hc  = 12;
constexpr int Ec  = 768;
constexpr int HSc = 64;
constexpr int Vc  = 32000;
constexpr int Bc  = 2;
constexpr int Tc  = 1024;
constexpr int Fc  = 4 * Ec;          // 3072
constexpr int M2c = Bc * Tc;         // 2048

// round-to-nearest-even f32 -> bf16
__device__ inline __bf16 f2bf(float f) {
    unsigned u = __float_as_uint(f);
    u += 0x7fffu + ((u >> 16) & 1u);
    unsigned short hs = (unsigned short)(u >> 16);
    union { unsigned short u16; __bf16 b; } cv; cv.u16 = hs;
    return cv.b;
}

// =================================================================
// bf16 WMMA GEMM with 2x2 register blocking:
//   C[M,N] = scale * A[M,K] * Bt[N,K]^T (+bias)(+res)(relu)
// A row-major [M,K] bf16, Bt row-major [N,K] bf16 (B transposed).
// Wave -> 32x32 output (2x2 subtiles of 16x16, 4 WMMAs / 4 frag loads
// per K-step of 32 -> 16 FLOP/byte from cache). Block = 4 waves (2x2)
// -> 64x64 tile. All M,N used are multiples of 64 (no edge handling).
// SMODE: 0 = f32 C[m*ldc+n]; 1 = bf16 C[m*ldc+n];
//        2 = bf16 qkv layout [(b*H+h)*T + t]*64 + k   (m=(b,t), n=(h,k))
//        3 = bf16 vT  layout [(b*H+h)*64 + k]*T + t
// CAUSAL : skip wave tiles strictly above diagonal (scores GEMM)
// CAUSALK: truncate K loop at row0+32 (wei @ V, rest of wei row is zero)
// =================================================================
template<int SMODE, bool BIAS, bool RELU, bool RES, bool CAUSAL, bool CAUSALK>
__global__ void __launch_bounds__(128)
gemm_wmma_bf16(const __bf16* __restrict__ A, int lda,
               const __bf16* __restrict__ Bt, int ldb,
               const float* __restrict__ bias,
               const float* __restrict__ res,
               void* __restrict__ Cp, int ldc,
               int M, int N, int K, float scale)
{
    const int lane  = threadIdx.x & 31;
    const int wave  = threadIdx.x >> 5;
    const int l16   = lane & 15;
    const int grp   = lane >> 4;                 // 0 or 1 (half-wave)
    const int waveR = wave >> 1;
    const int waveC = wave & 1;
    const int row0  = blockIdx.y * 64 + waveR * 32;
    const int col0  = blockIdx.x * 64 + waveC * 32;
    if (col0 >= N || row0 >= M) return;
    if (CAUSAL && col0 >= row0 + 32) return;     // fully masked wave tile

    int kend = K;
    if (CAUSALK) { int ke = row0 + 32; kend = (ke < K) ? ke : K; }

    // A fragment: lane holds row M=l16; elems 0..7 -> K = 8*grp + 0..7,
    //             elems 8..15 -> K = 8*grp + 16..23  (ISA 16-bit A 16x32 layout)
    const __bf16* Arow0 = A + (size_t)(row0      + l16) * lda;
    const __bf16* Arow1 = A + (size_t)(row0 + 16 + l16) * lda;
    // B fragment: lane holds column N=l16; elems 0..7 -> K = 8*grp + 0..7, elems 8..15 -> K = 8*grp + 16..23
    const __bf16* Bcol0 = Bt + (size_t)(col0      + l16) * ldb;
    const __bf16* Bcol1 = Bt + (size_t)(col0 + 16 + l16) * ldb;

    v8f acc00 = {0.f,0.f,0.f,0.f,0.f,0.f,0.f,0.f};
    v8f acc01 = acc00, acc10 = acc00, acc11 = acc00;

    for (int k0 = 0; k0 < kend; k0 += 32) {
        V16U a0, a1, b0, b1;
        a0.h[0] = *(const v8bf*)(Arow0 + k0 + 8 * grp);
        a0.h[1] = *(const v8bf*)(Arow0 + k0 + 16 + 8 * grp);
        a1.h[0] = *(const v8bf*)(Arow1 + k0 + 8 * grp);
        a1.h[1] = *(const v8bf*)(Arow1 + k0 + 16 + 8 * grp);
        b0.h[0] = *(const v8bf*)(Bcol0 + k0 + 8 * grp);
        b0.h[1] = *(const v8bf*)(Bcol0 + k0 + 16 + 8 * grp);
        b1.h[0] = *(const v8bf*)(Bcol1 + k0 + 8 * grp);
        b1.h[1] = *(const v8bf*)(Bcol1 + k0 + 16 + 8 * grp);
        acc00 = __builtin_amdgcn_wmma_f32_16x16x32_bf16(
            false, a0.v, false, b0.v, (short)0, acc00, false, false);
        acc01 = __builtin_amdgcn_wmma_f32_16x16x32_bf16(
            false, a0.v, false, b1.v, (short)0, acc01, false, false);
        acc10 = __builtin_amdgcn_wmma_f32_16x16x32_bf16(
            false, a1.v, false, b0.v, (short)0, acc10, false, false);
        acc11 = __builtin_amdgcn_wmma_f32_16x16x32_bf16(
            false, a1.v, false, b1.v, (short)0, acc11, false, false);
    }

    // C/D layout: VGPR r, lanes 0-15 -> M=r, lanes 16-31 -> M=8+r; N = lane%16
    v8f* accs[4] = { &acc00, &acc01, &acc10, &acc11 };
#pragma unroll
    for (int s = 0; s < 4; ++s) {
        const int mbase = row0 + (s >> 1) * 16;
        const int n     = col0 + (s & 1) * 16 + l16;
        const v8f& acc  = *accs[s];
#pragma unroll
        for (int r = 0; r < 8; ++r) {
            const int m = mbase + r + 8 * grp;
            float v = acc[r] * scale;
            if constexpr (BIAS) v += bias[n];
            if constexpr (RES)  v += res[(size_t)m * ldc + n];
            if constexpr (RELU) v = v > 0.f ? v : 0.f;
            if constexpr (SMODE == 0) {
                ((float*)Cp)[(size_t)m * ldc + n] = v;
            } else if constexpr (SMODE == 1) {
                ((__bf16*)Cp)[(size_t)m * ldc + n] = f2bf(v);
            } else if constexpr (SMODE == 2) {
                const int bb = m >> 10, tt = m & (Tc - 1);
                const int hh = n >> 6,  kk = n & (HSc - 1);
                ((__bf16*)Cp)[(((size_t)(bb * Hc + hh) * Tc) + tt) * HSc + kk] = f2bf(v);
            } else {  // SMODE == 3 : store V transposed [.., HS, T]
                const int bb = m >> 10, tt = m & (Tc - 1);
                const int hh = n >> 6,  kk = n & (HSc - 1);
                ((__bf16*)Cp)[(((size_t)(bb * Hc + hh) * HSc) + kk) * Tc + tt] = f2bf(v);
            }
        }
    }
}

// ---------------- embedding: x = tok_emb[idx] + pos_emb[t] ----------------
__global__ void embed_k(const int* __restrict__ idx, const float* __restrict__ te,
                        const float* __restrict__ pe, float* __restrict__ x)
{
    int i = blockIdx.x * 256 + threadIdx.x;
    if (i >= M2c * Ec) return;
    int m = i / Ec, e = i - m * Ec;
    int t = m & (Tc - 1);
    x[i] = te[(size_t)idx[m] * Ec + e] + pe[(size_t)t * Ec + e];
}

// ---------------- LayerNorm (row of 768) -> bf16 ----------------
__global__ void layernorm_bf16(const float* __restrict__ x, const float* __restrict__ g,
                               const float* __restrict__ b, __bf16* __restrict__ o)
{
    __shared__ float red[256];
    const int row = blockIdx.x, tid = threadIdx.x;
    const float* xr = x + (size_t)row * Ec;
    float ls = 0.f, lq = 0.f;
    for (int e = tid; e < Ec; e += 256) { float v = xr[e]; ls += v; lq += v * v; }
    red[tid] = ls; __syncthreads();
    for (int s = 128; s > 0; s >>= 1) { if (tid < s) red[tid] += red[tid + s]; __syncthreads(); }
    const float mean = red[0] * (1.f / Ec);
    __syncthreads();
    red[tid] = lq; __syncthreads();
    for (int s = 128; s > 0; s >>= 1) { if (tid < s) red[tid] += red[tid + s]; __syncthreads(); }
    const float var  = red[0] * (1.f / Ec) - mean * mean;
    const float rstd = rsqrtf(var + 1e-5f);
    __bf16* orow = o + (size_t)row * Ec;
    for (int e = tid; e < Ec; e += 256)
        orow[e] = f2bf((xr[e] - mean) * rstd * g[e] + b[e]);
}

// ---------------- causal softmax: f32 scores row -> bf16 weights ----------------
__global__ void softmax_causal(const float* __restrict__ sc, __bf16* __restrict__ w)
{
    __shared__ float red[256];
    const int row = blockIdx.x, tid = threadIdx.x;
    const float* srow = sc + (size_t)row * Tc;
    float mx = -__builtin_inff();
    for (int s = tid; s <= row; s += 256) mx = fmaxf(mx, srow[s]);
    red[tid] = mx; __syncthreads();
    for (int s = 128; s > 0; s >>= 1) { if (tid < s) red[tid] = fmaxf(red[tid], red[tid + s]); __syncthreads(); }
    mx = red[0]; __syncthreads();
    float sm = 0.f;
    for (int s = tid; s <= row; s += 256) sm += expf(srow[s] - mx);
    red[tid] = sm; __syncthreads();
    for (int s = 128; s > 0; s >>= 1) { if (tid < s) red[tid] += red[tid + s]; __syncthreads(); }
    const float inv = 1.f / red[0];
    __bf16* wrow = w + (size_t)row * Tc;
    for (int s = tid; s < Tc; s += 256)
        wrow[s] = (s <= row) ? f2bf(expf(srow[s] - mx) * inv) : f2bf(0.f);
}

// ---------------- per-row -log p(target) over V=32000 ----------------
__global__ void row_logloss(const float* __restrict__ logits, const int* __restrict__ tgt,
                            float* __restrict__ rl)
{
    __shared__ float red[256];
    const int row = blockIdx.x, tid = threadIdx.x;
    const float* lr = logits + (size_t)row * Vc;
    float mx = -__builtin_inff();
    for (int v = tid; v < Vc; v += 256) mx = fmaxf(mx, lr[v]);
    red[tid] = mx; __syncthreads();
    for (int s = 128; s > 0; s >>= 1) { if (tid < s) red[tid] = fmaxf(red[tid], red[tid + s]); __syncthreads(); }
    mx = red[0]; __syncthreads();
    float sm = 0.f;
    for (int v = tid; v < Vc; v += 256) sm += expf(lr[v] - mx);
    red[tid] = sm; __syncthreads();
    for (int s = 128; s > 0; s >>= 1) { if (tid < s) red[tid] += red[tid + s]; __syncthreads(); }
    if (tid == 0) rl[row] = -(lr[tgt[row]] - mx - logf(red[0]));
}

__global__ void loss_mean(const float* __restrict__ rl, float* __restrict__ out)
{
    __shared__ float red[256];
    const int tid = threadIdx.x;
    float s = 0.f;
    for (int i = tid; i < M2c; i += 256) s += rl[i];
    red[tid] = s; __syncthreads();
    for (int st = 128; st > 0; st >>= 1) { if (tid < st) red[tid] += red[tid + st]; __syncthreads(); }
    if (tid == 0) out[0] = red[0] * (1.f / M2c);
}

// ---------------- weight repacks (f32 -> bf16, to [N,K]) ----------------
// Wq/Wk/Wv [L,H,E,HS] -> Wt [L, n=(h*64+k), e]
__global__ void repack_qkv(const float* __restrict__ W, __bf16* __restrict__ Wt)
{
    const int tot = Lc * Ec * Ec;
    int i = blockIdx.x * 256 + threadIdx.x;
    if (i >= tot) return;
    int l = i / (Ec * Ec); int rem = i - l * (Ec * Ec);
    int n = rem / Ec;      int e = rem - n * Ec;
    int h = n >> 6, k = n & 63;
    Wt[i] = f2bf(W[(((size_t)(l * Hc + h) * Ec) + e) * HSc + k]);
}

// W [count, rows, cols] -> Wt [count, cols, rows]
__global__ void repack_T(const float* __restrict__ W, __bf16* __restrict__ Wt,
                         int rows, int cols, int total)
{
    int i = blockIdx.x * 256 + threadIdx.x;
    if (i >= total) return;
    const int per = rows * cols;
    int l = i / per; int rem = i - l * per;
    int c = rem / rows; int r = rem - c * rows;
    Wt[i] = f2bf(W[(size_t)l * per + (size_t)r * cols + c]);
}

// =================================================================
extern "C" void kernel_launch(void* const* d_in, const int* in_sizes, int n_in,
                              void* d_out, int out_size, void* d_ws, size_t ws_size,
                              hipStream_t stream)
{
    (void)in_sizes; (void)n_in; (void)out_size; (void)ws_size;

    const int*   idx     = (const int*)  d_in[0];
    const int*   targets = (const int*)  d_in[1];
    const float* tok_emb = (const float*)d_in[2];
    const float* pos_emb = (const float*)d_in[3];
    const float* Wq  = (const float*)d_in[4];  const float* bq  = (const float*)d_in[5];
    const float* Wk  = (const float*)d_in[6];  const float* bk  = (const float*)d_in[7];
    const float* Wv  = (const float*)d_in[8];  const float* bv  = (const float*)d_in[9];
    const float* Wo  = (const float*)d_in[10]; const float* bo  = (const float*)d_in[11];
    const float* ln1g= (const float*)d_in[12]; const float* ln1b= (const float*)d_in[13];
    const float* W1  = (const float*)d_in[14]; const float* b1  = (const float*)d_in[15];
    const float* W2  = (const float*)d_in[16]; const float* b2  = (const float*)d_in[17];
    const float* ln2g= (const float*)d_in[18]; const float* ln2b= (const float*)d_in[19];
    const float* lnfg= (const float*)d_in[20]; const float* lnfb= (const float*)d_in[21];
    const float* Wlm = (const float*)d_in[22]; const float* blm = (const float*)d_in[23];

    // ---- workspace carving (256B aligned) ----
    char* wsb = (char*)d_ws;
    size_t cur = 0;
    auto carve = [&](size_t bytes) -> void* {
        void* p = wsb + cur;
        cur = (cur + bytes + 255) & ~(size_t)255;
        return p;
    };
    __bf16* Wqt  = (__bf16*)carve((size_t)Lc * Ec * Ec * 2);
    __bf16* Wkt  = (__bf16*)carve((size_t)Lc * Ec * Ec * 2);
    __bf16* Wvt  = (__bf16*)carve((size_t)Lc * Ec * Ec * 2);
    __bf16* Wot  = (__bf16*)carve((size_t)Lc * Ec * Ec * 2);
    __bf16* W1t  = (__bf16*)carve((size_t)Lc * Ec * Fc * 2);
    __bf16* W2t  = (__bf16*)carve((size_t)Lc * Fc * Ec * 2);
    __bf16* Wlmt = (__bf16*)carve((size_t)Ec * Vc * 2);
    float*  xb   = (float*) carve((size_t)M2c * Ec * 4);
    __bf16* hb   = (__bf16*)carve((size_t)M2c * Ec * 2);
    __bf16* qb   = (__bf16*)carve((size_t)M2c * Ec * 2);
    __bf16* kb   = (__bf16*)carve((size_t)M2c * Ec * 2);
    __bf16* vtb  = (__bf16*)carve((size_t)M2c * Ec * 2);
    float*  scb  = (float*) carve((size_t)Tc * Tc * 4);
    __bf16* weib = (__bf16*)carve((size_t)Tc * Tc * 2);
    __bf16* ob   = (__bf16*)carve((size_t)M2c * Ec * 2);
    __bf16* h1b  = (__bf16*)carve((size_t)M2c * Fc * 2);
    __bf16* hfb  = (__bf16*)carve((size_t)M2c * Ec * 2);
    float*  rlb  = (float*) carve((size_t)M2c * 4);

    float* logits = (float*)d_out;
    float* lossp  = logits + (size_t)M2c * Vc;

    auto blks = [](int n) { return dim3((n + 255) / 256); };
    auto gg   = [](int N, int M) { return dim3(N / 64, M / 64); };  // 64x64 block tiles
    const dim3 gb(128);

    // ---- repack weights to bf16 [N,K] ----
    {
        const int nq = Lc * Ec * Ec;
        repack_qkv<<<blks(nq), 256, 0, stream>>>(Wq, Wqt);
        repack_qkv<<<blks(nq), 256, 0, stream>>>(Wk, Wkt);
        repack_qkv<<<blks(nq), 256, 0, stream>>>(Wv, Wvt);
        repack_T<<<blks(Lc * Ec * Ec), 256, 0, stream>>>(Wo,  Wot,  Ec, Ec, Lc * Ec * Ec);
        repack_T<<<blks(Lc * Ec * Fc), 256, 0, stream>>>(W1,  W1t,  Ec, Fc, Lc * Ec * Fc);
        repack_T<<<blks(Lc * Fc * Ec), 256, 0, stream>>>(W2,  W2t,  Fc, Ec, Lc * Fc * Ec);
        repack_T<<<blks(Ec * Vc),      256, 0, stream>>>(Wlm, Wlmt, Ec, Vc, Ec * Vc);
    }

    // ---- embeddings ----
    embed_k<<<blks(M2c * Ec), 256, 0, stream>>>(idx, tok_emb, pos_emb, xb);

    const float att_scale = 0.125f;   // 64^-0.5

    for (int l = 0; l < Lc; ++l) {
        const size_t wofE = (size_t)l * Ec * Ec;
        // ln1
        layernorm_bf16<<<M2c, 256, 0, stream>>>(xb, ln1g + l * Ec, ln1b + l * Ec, hb);
        // q, k, v projections (q/k per-head layout, v transposed per head)
        gemm_wmma_bf16<2, true, false, false, false, false><<<gg(Ec, M2c), gb, 0, stream>>>(
            hb, Ec, Wqt + wofE, Ec, bq + l * Ec, nullptr, qb, 0, M2c, Ec, Ec, 1.f);
        gemm_wmma_bf16<2, true, false, false, false, false><<<gg(Ec, M2c), gb, 0, stream>>>(
            hb, Ec, Wkt + wofE, Ec, bk + l * Ec, nullptr, kb, 0, M2c, Ec, Ec, 1.f);
        gemm_wmma_bf16<3, true, false, false, false, false><<<gg(Ec, M2c), gb, 0, stream>>>(
            hb, Ec, Wvt + wofE, Ec, bv + l * Ec, nullptr, vtb, 0, M2c, Ec, Ec, 1.f);

        // attention per (b, head), reusing the [T,T] score buffer
        for (int bh = 0; bh < Bc * Hc; ++bh) {
            const int bI = bh / Hc, hI = bh % Hc;
            const __bf16* qp = qb  + (size_t)bh * Tc * HSc;
            const __bf16* kp = kb  + (size_t)bh * Tc * HSc;
            const __bf16* vp = vtb + (size_t)bh * HSc * Tc;
            // scores = scale * q @ k^T (causal tiles only)
            gemm_wmma_bf16<0, false, false, false, true, false><<<gg(Tc, Tc), gb, 0, stream>>>(
                qp, HSc, kp, HSc, nullptr, nullptr, scb, Tc, Tc, Tc, HSc, att_scale);
            softmax_causal<<<Tc, 256, 0, stream>>>(scb, weib);
            // o = wei @ v  (K truncated at diagonal); write into concat-head buffer
            gemm_wmma_bf16<1, false, false, false, false, true><<<gg(HSc, Tc), gb, 0, stream>>>(
                weib, Tc, vp, Tc, nullptr, nullptr,
                ob + (size_t)bI * Tc * Ec + (size_t)hI * HSc, Ec, Tc, HSc, Tc, 1.f);
        }

        // x = x + o @ Wo + bo
        gemm_wmma_bf16<0, true, false, true, false, false><<<gg(Ec, M2c), gb, 0, stream>>>(
            ob, Ec, Wot + wofE, Ec, bo + l * Ec, xb, xb, Ec, M2c, Ec, Ec, 1.f);
        // ln2
        layernorm_bf16<<<M2c, 256, 0, stream>>>(xb, ln2g + l * Ec, ln2b + l * Ec, hb);
        // h1 = relu(h @ W1 + b1)
        gemm_wmma_bf16<1, true, true, false, false, false><<<gg(Fc, M2c), gb, 0, stream>>>(
            hb, Ec, W1t + (size_t)l * Ec * Fc, Ec, b1 + l * Fc, nullptr, h1b, Fc, M2c, Fc, Ec, 1.f);
        // x = x + h1 @ W2 + b2
        gemm_wmma_bf16<0, true, false, true, false, false><<<gg(Ec, M2c), gb, 0, stream>>>(
            h1b, Fc, W2t + (size_t)l * Fc * Ec, Fc, b2 + l * Ec, xb, xb, Ec, M2c, Ec, Fc, 1.f);
    }

    // final LN + LM head
    layernorm_bf16<<<M2c, 256, 0, stream>>>(xb, lnfg, lnfb, hfb);
    gemm_wmma_bf16<0, true, false, false, false, false><<<gg(Vc, M2c), gb, 0, stream>>>(
        hfb, Ec, Wlmt, Ec, blm, nullptr, logits, Vc, M2c, Vc, Ec, 1.f);

    // loss = mean(-log_softmax(logits)[target])
    row_logloss<<<M2c, 256, 0, stream>>>(logits, targets, rlb);
    loss_mean<<<1, 256, 0, stream>>>(rlb, lossp);
}
